// SelectiveQuantizedMultiHeadAttention_47184510714570
// MI455X (gfx1250) — 16-bit and 8-bit matrix operand order repaired and checked by reading, compiled, not run on hardware
//
#include <hip/hip_runtime.h>
#include <hip/hip_bf16.h>
#include <stdint.h>


typedef __attribute__((ext_vector_type(16))) _Float16 v16h;
typedef __attribute__((ext_vector_type(8)))  float    v8f;
typedef __attribute__((ext_vector_type(8)))  int      v8i;
typedef __attribute__((ext_vector_type(4)))  int      v4i;
typedef __attribute__((ext_vector_type(2)))  int      v2i;
typedef __attribute__((ext_vector_type(8)))  unsigned v8u;
typedef __attribute__((ext_vector_type(4)))  unsigned v4u;

struct H16 { _Float16 h[16]; };

#if defined(__gfx1250__)
#define USE_ASYNC_LDS 1
#else
#define USE_ASYNC_LDS 0
#endif

#if USE_ASYNC_LDS
static __device__ inline void async_ld_lds_b128(const void* gp, void* lp) {
  unsigned lds = (unsigned)(unsigned long long)lp;
  unsigned long long ga = (unsigned long long)gp;
  asm volatile("global_load_async_to_lds_b128 %0, %1, off"
               :: "v"(lds), "v"(ga) : "memory");
}
static __device__ inline void wait_async0() {
#if __has_builtin(__builtin_amdgcn_s_wait_asynccnt)
  __builtin_amdgcn_s_wait_asynccnt(0);
#else
  asm volatile("s_wait_asynccnt 0x0" ::: "memory");
#endif
}
#endif

static __device__ inline unsigned enc_f(float f) {
  unsigned u = __float_as_uint(f);
  return (u >> 31) ? ~u : (u | 0x80000000u);
}
static __device__ inline float dec_f(unsigned u) {
  return (u >> 31) ? __uint_as_float(u & 0x7FFFFFFFu) : __uint_as_float(~u);
}

static __device__ inline v8i lda8(const int8_t* p) {
  v2i c0 = *(const v2i*)(p), c1 = *(const v2i*)(p + 16);
  v2i c2 = *(const v2i*)(p + 32), c3 = *(const v2i*)(p + 48);
  v8i a;
  a[0] = c0[0]; a[1] = c0[1]; a[2] = c1[0]; a[3] = c1[1];
  a[4] = c2[0]; a[5] = c2[1]; a[6] = c3[0]; a[7] = c3[1];
  return a;
}
static __device__ inline v8i ldb8(const int8_t* p) {
  v4i c0 = *(const v4i*)(p), c1 = *(const v4i*)(p + 32);
  v8i b;
#pragma unroll
  for (int e = 0; e < 4; ++e) { b[e] = c0[e]; b[4 + e] = c1[e]; }
  return b;
}
static __device__ inline v16h ldh16(const _Float16* p) {
  return __builtin_bit_cast(v16h, *(const v8u*)p);
}
static __device__ inline v16h ldh16_2(const _Float16* p0, const _Float16* p1) {
  v4u lo = *(const v4u*)p0, hi = *(const v4u*)p1;
  v8u t;
#pragma unroll
  for (int e = 0; e < 4; ++e) { t[e] = lo[e]; t[4 + e] = hi[e]; }
  return __builtin_bit_cast(v16h, t);
}

__global__ void init_minmax_k(unsigned* mm) {
  int t = threadIdx.x;
  if (t < 8) mm[t] = (t & 1) ? 0u : 0xFFFFFFFFu;
}

__global__ __launch_bounds__(256) void reduce_minmax_k(const float* __restrict__ src,
                                                       int n, unsigned* mm) {
  __shared__ unsigned smin[256], smax[256];
  unsigned lmin = 0xFFFFFFFFu, lmax = 0u;
  for (int i = blockIdx.x * 256 + threadIdx.x; i < n; i += gridDim.x * 256) {
    unsigned e = enc_f(src[i]);
    lmin = min(lmin, e); lmax = max(lmax, e);
  }
  int t = threadIdx.x;
  smin[t] = lmin; smax[t] = lmax;
  __syncthreads();
  for (int off = 128; off > 0; off >>= 1) {
    if (t < off) { smin[t] = min(smin[t], smin[t + off]); smax[t] = max(smax[t], smax[t + off]); }
    __syncthreads();
  }
  if (t == 0) { atomicMin(&mm[0], smin[0]); atomicMax(&mm[1], smax[0]); }
}

__global__ void compute_params_k(const unsigned* mm, float* params) {
  int s = threadIdx.x;
  if (s < 4) {
    float mn = fminf(dec_f(mm[2 * s]), 0.0f);
    float mx = fmaxf(dec_f(mm[2 * s + 1]), 0.0f);
    float sc = fmaxf((mx - mn) / 255.0f, 1e-8f);
    float zp = fminf(fmaxf(rintf(-128.0f - mn / sc), -128.0f), 127.0f);
    params[2 * s] = sc; params[2 * s + 1] = zp;
  }
}

__global__ __launch_bounds__(256) void quant_rows_k(const float* __restrict__ src,
                                                    const float* __restrict__ params,
                                                    int slot, int8_t* __restrict__ dst,
                                                    int* __restrict__ rsum) {
  const int row = blockIdx.x, t = threadIdx.x;
  const float s = params[2 * slot], zp = params[2 * slot + 1];
  const float* r = src + (size_t)row * 1024;
  int8_t* d = dst + (size_t)row * 1024;
  int acc = 0;
#pragma unroll
  for (int i = t; i < 1024; i += 256) {
    float q = fminf(fmaxf(rintf(r[i] / s) + zp, -128.0f), 127.0f);
    int qi = (int)q;
    d[i] = (int8_t)qi;
    acc += qi;
  }
  __shared__ int red[256];
  red[t] = acc;
  __syncthreads();
  for (int off = 128; off > 0; off >>= 1) {
    if (t < off) red[t] += red[t + off];
    __syncthreads();
  }
  if (t == 0) rsum[row] = red[0];
}

__global__ __launch_bounds__(128) void gemm_q8_k(
    const int8_t* __restrict__ A, const int8_t* __restrict__ W,
    const int* __restrict__ rsumA, const int* __restrict__ rsumW,
    const float* __restrict__ params, int pa, int pw, int Nout,
    _Float16* __restrict__ outH, float* __restrict__ outF) {
  const int K = 1024;
  const int lane = threadIdx.x & 31, wv = threadIdx.x >> 5;
  const int l15 = lane & 15, lhi = lane >> 4;
  const int m0 = blockIdx.x * 64 + (wv & 1) * 32;
  const int o0 = blockIdx.y * 128 + (wv >> 1) * 64;

  v8i acc[2][4];
#pragma unroll
  for (int i = 0; i < 2; ++i)
#pragma unroll
    for (int j = 0; j < 4; ++j)
#pragma unroll
      for (int e = 0; e < 8; ++e) acc[i][j][e] = 0;

  for (int k = 0; k < K; k += 64) {
    v8i a[2], b[4];
#pragma unroll
    for (int i = 0; i < 2; ++i)
      a[i] = lda8(A + (size_t)(m0 + 16 * i + l15) * K + k + lhi * 8);
#pragma unroll
    for (int j = 0; j < 4; ++j)
      b[j] = lda8(W + (size_t)(o0 + 16 * j + l15) * K + k + lhi * 8);
#pragma unroll
    for (int i = 0; i < 2; ++i)
#pragma unroll
      for (int j = 0; j < 4; ++j)
        acc[i][j] = __builtin_amdgcn_wmma_i32_16x16x64_iu8(true, a[i], true, b[j],
                                                           acc[i][j], false, false);
  }

  const float sA = params[2 * pa], zA = params[2 * pa + 1];
  const float sW = params[2 * pw], zW = params[2 * pw + 1];
  const float ss = sA * sW, kzz = 1024.0f * zA * zW;

  float rA[2][8];
#pragma unroll
  for (int i = 0; i < 2; ++i)
#pragma unroll
    for (int v = 0; v < 8; ++v)
      rA[i][v] = (float)rsumA[m0 + 16 * i + v + lhi * 8];

#pragma unroll
  for (int j = 0; j < 4; ++j) {
    const int oc = o0 + 16 * j + l15;
    const float rw = (float)rsumW[oc];
#pragma unroll
    for (int i = 0; i < 2; ++i) {
#pragma unroll
      for (int v = 0; v < 8; ++v) {
        const size_t m = (size_t)(m0 + 16 * i + v + lhi * 8);
        const float val = ss * ((float)acc[i][j][v] - zA * rw - zW * rA[i][v] + kzz);
        if (outH) outH[m * (size_t)Nout + oc] = (_Float16)val;
        else      outF[m * (size_t)Nout + oc] = val;
      }
    }
  }
}

__global__ __launch_bounds__(128) void attn_k(const _Float16* __restrict__ qkv,
                                              float* __restrict__ out) {
  const int id = blockIdx.x;
  const int qt = id & 15;
  const int h  = (id >> 4) & 15;
  const int b  = id >> 8;
  const int tid = threadIdx.x;
  const int lane = tid & 31, w = tid >> 5;
  const int l15 = lane & 15, lhi = lane >> 4;
  const size_t RS = 3072;

  const _Float16* Qg = qkv + (size_t)(b * 1024 + qt * 64) * RS + h * 64;
  const _Float16* Kg = qkv + (size_t)(b * 1024) * RS + 1024 + h * 64;
  const _Float16* Vg = Kg + 1024;

  __shared__ __align__(32) _Float16 Kt[64][64];
  __shared__ __align__(32) _Float16 Vt[64][64];
  __shared__ float Srow[4][16][64];
  __shared__ __align__(32) _Float16 Pst[4][16][64];
  __shared__ float mrun[4][16], lrun[4][16], rscale[4][16];

  v16h aq[2];
  {
    const _Float16* qp = Qg + (size_t)(16 * w + l15) * RS;
#pragma unroll
    for (int kk = 0; kk < 2; ++kk) {
      const int d0 = 32 * kk + lhi * 16;
      aq[kk] = ldh16_2(qp + d0, qp + d0 + 8);
    }
  }

  v8f o[4];
#pragma unroll
  for (int t = 0; t < 4; ++t)
#pragma unroll
    for (int e = 0; e < 8; ++e) o[t][e] = 0.0f;

  if (lane < 16) { mrun[w][lane] = -1e30f; lrun[w][lane] = 0.0f; }

  for (int j = 0; j < 16; ++j) {
    __syncthreads();
    {
      const _Float16* kp = Kg + (size_t)(64 * j + (tid >> 1)) * RS + (tid & 1) * 32;
      _Float16* lp = &Kt[tid >> 1][(tid & 1) * 32];
#if USE_ASYNC_LDS
#pragma unroll
      for (int c = 0; c < 4; ++c)
        async_ld_lds_b128(kp + 8 * c, lp + 8 * c);
#else
      v8u k0 = *(const v8u*)kp;
      v8u k1 = *(const v8u*)(kp + 16);
      *(v8u*)lp = k0;
      *(v8u*)(lp + 16) = k1;
#endif
    }
#pragma unroll
    for (int it = 0; it < 2; ++it) {
      const int c = tid + 128 * it;
      const int kv = c >> 2, dc = c & 3;
      const _Float16* vp = Vg + (size_t)(64 * j + kv) * RS + 16 * dc;
      H16 hh = __builtin_bit_cast(H16, *(const v8u*)vp);
#pragma unroll
      for (int e = 0; e < 16; ++e) Vt[16 * dc + e][kv] = hh.h[e];
    }
#if USE_ASYNC_LDS
    wait_async0();
#endif
    __syncthreads();

#pragma unroll
    for (int n = 0; n < 4; ++n) {
      v8f c;
#pragma unroll
      for (int e = 0; e < 8; ++e) c[e] = 0.0f;
#pragma unroll
      for (int kk = 0; kk < 2; ++kk) {
        v16h bk = ldh16(&Kt[16 * n + l15][32 * kk + lhi * 16]);
        c = __builtin_amdgcn_wmma_f32_16x16x32_f16(false, aq[kk], false, bk,
                                                   (short)0, c, false, false);
      }
#pragma unroll
      for (int v = 0; v < 8; ++v)
        Srow[w][v + lhi * 8][16 * n + l15] = c[v];
    }

    if (lane < 16) {
      const int r = lane;
      float mo = mrun[w][r], mx = mo;
      for (int cc = 0; cc < 64; ++cc) mx = fmaxf(mx, Srow[w][r][cc] * 0.125f);
      float alpha = __expf(mo - mx);
      float sum = 0.0f;
      for (int cc = 0; cc < 64; ++cc) {
        float p = __expf(Srow[w][r][cc] * 0.125f - mx);
        Pst[w][r][cc] = (_Float16)p;
        sum += p;
      }
      lrun[w][r] = lrun[w][r] * alpha + sum;
      mrun[w][r] = mx;
      rscale[w][r] = alpha;
    }

    float rs[8];
#pragma unroll
    for (int v = 0; v < 8; ++v) rs[v] = rscale[w][v + lhi * 8];
#pragma unroll
    for (int t = 0; t < 4; ++t)
#pragma unroll
      for (int v = 0; v < 8; ++v) o[t][v] *= rs[v];

    v16h pa[2];
#pragma unroll
    for (int kk = 0; kk < 2; ++kk) {
      const _Float16* pp = &Pst[w][l15][32 * kk + lhi * 16];
      pa[kk] = ldh16_2(pp, pp + 8);
    }
#pragma unroll
    for (int t = 0; t < 4; ++t) {
#pragma unroll
      for (int kk = 0; kk < 2; ++kk) {
        v16h bv = ldh16(&Vt[16 * t + l15][32 * kk + lhi * 16]);
        o[t] = __builtin_amdgcn_wmma_f32_16x16x32_f16(false, pa[kk], false, bv,
                                                      (short)0, o[t], false, false);
      }
    }
  }

  float li[8];
#pragma unroll
  for (int v = 0; v < 8; ++v) li[v] = 1.0f / lrun[w][v + lhi * 8];
#pragma unroll
  for (int t = 0; t < 4; ++t)
#pragma unroll
    for (int v = 0; v < 8; ++v) {
      const int row = qt * 64 + 16 * w + v + lhi * 8;
      out[(size_t)(b * 1024 + row) * 1024 + h * 64 + 16 * t + l15] = o[t][v] * li[v];
    }
}

extern "C" void kernel_launch(void* const* d_in, const int* in_sizes, int n_in,
                              void* d_out, int out_size, void* d_ws, size_t ws_size,
                              hipStream_t stream) {
  (void)in_sizes; (void)n_in; (void)out_size; (void)ws_size;
  const float* x      = (const float*)d_in[0];
  const float* w_qkv  = (const float*)d_in[1];
  const float* w_proj = (const float*)d_in[2];
  float* out = (float*)d_out;

  char* ws = (char*)d_ws;
  size_t off = 0;
  auto take = [&](size_t bytes) -> char* {
    char* p = ws + off;
    off = (off + bytes + 255) & ~(size_t)255;
    return p;
  };
  unsigned* minmax = (unsigned*)take(8 * sizeof(unsigned));
  float*    params = (float*)take(8 * sizeof(float));
  int* rsx  = (int*)take(8192 * 4);
  int* rsw1 = (int*)take(3072 * 4);
  int* rsw2 = (int*)take(1024 * 4);
  int* rso  = (int*)take(8192 * 4);
  int8_t* qx  = (int8_t*)take((size_t)8192 * 1024);
  int8_t* qw1 = (int8_t*)take((size_t)3072 * 1024);
  int8_t* qw2 = (int8_t*)take((size_t)1024 * 1024);
  int8_t* qo  = (int8_t*)take((size_t)8192 * 1024);
  _Float16* qkvh = (_Float16*)take((size_t)8192 * 3072 * 2);
  float* attn = (float*)take((size_t)8192 * 1024 * 4);

  init_minmax_k<<<1, 32, 0, stream>>>(minmax);
  reduce_minmax_k<<<512, 256, 0, stream>>>(x, 8 * 1024 * 1024, minmax + 0);
  reduce_minmax_k<<<256, 256, 0, stream>>>(w_qkv, 3 * 1024 * 1024, minmax + 2);
  reduce_minmax_k<<<128, 256, 0, stream>>>(w_proj, 1024 * 1024, minmax + 4);
  compute_params_k<<<1, 32, 0, stream>>>(minmax, params);

  quant_rows_k<<<8192, 256, 0, stream>>>(x, params, 0, qx, rsx);
  quant_rows_k<<<3072, 256, 0, stream>>>(w_qkv, params, 1, qw1, rsw1);
  quant_rows_k<<<1024, 256, 0, stream>>>(w_proj, params, 2, qw2, rsw2);

  gemm_q8_k<<<dim3(128, 24), 128, 0, stream>>>(qx, qw1, rsx, rsw1, params, 0, 1,
                                               3072, qkvh, nullptr);

  attn_k<<<2048, 128, 0, stream>>>(qkvh, attn);

  reduce_minmax_k<<<512, 256, 0, stream>>>(attn, 8 * 1024 * 1024, minmax + 6);
  compute_params_k<<<1, 32, 0, stream>>>(minmax, params);
  quant_rows_k<<<8192, 256, 0, stream>>>(attn, params, 3, qo, rso);

  gemm_q8_k<<<dim3(128, 8), 128, 0, stream>>>(qo, qw2, rso, rsw2, params, 3, 2,
                                              1024, nullptr, out);
}
